// CausalSelfAttention_29592324669822
// MI455X (gfx1250) — matrix operand order repaired and checked by reading, not compiled, not run on hardware
//
#include <hip/hip_runtime.h>

// ---------------------------------------------------------------------------
// Causal self-attention (B=8, T=1024, C=1024, H=16, hd=64) for gfx1250.
// bf16 WMMA (v_wmma_f32_16x16x32_bf16) everywhere; flash-attention streaming
// softmax so the 512MB T*T score matrix is never materialized; all
// intermediates live in d_ws (~88MB, L2-resident on MI455X's 192MB L2).
// ---------------------------------------------------------------------------

typedef __attribute__((ext_vector_type(16))) __bf16 v16bf;
typedef __attribute__((ext_vector_type(8)))  __bf16 v8bf;
typedef __attribute__((ext_vector_type(8)))  float  v8f;
typedef unsigned short ushort_t;

#define BATCH   8
#define SEQ     1024
#define EMB     1024
#define N_HEAD  16
#define HEAD_DIM 64
#define QKV_N   3072

static __device__ __forceinline__ ushort_t f2bf(float f) {
  unsigned int u = __float_as_uint(f);
  u += 0x7FFFu + ((u >> 16) & 1u);   // round-to-nearest-even
  return (ushort_t)(u >> 16);
}

static __device__ __forceinline__ v8f zero8() {
  v8f z;
#pragma unroll
  for (int i = 0; i < 8; ++i) z[i] = 0.0f;
  return z;
}

static __device__ __forceinline__ v16bf combine16(v8bf lo, v8bf hi) {
  v16bf r;
#pragma unroll
  for (int i = 0; i < 8; ++i) { r[i] = lo[i]; r[i + 8] = hi[i]; }
  return r;
}

static __device__ __forceinline__ v8f wmma_bf16(v16bf a, v16bf b, v8f c) {
  // (neg_a, A, neg_b, B, c_mod, C, reuse_a, reuse_b)
  return __builtin_amdgcn_wmma_f32_16x16x32_bf16(false, a, false, b,
                                                 (short)0, c, false, false);
}

// A fragment (16x32 bf16, row-major source [M][K]):
// lane m = lane&15; lane-half selects K blocks {0..7,16..23} vs {8..15,24..31}.
static __device__ __forceinline__ v16bf load_a_frag(const ushort_t* base,
                                                    int rowStride, int row,
                                                    int k, int lane) {
  const ushort_t* p =
      base + (size_t)(row + (lane & 15)) * rowStride + k + ((lane >> 4) * 8);
  v8bf lo = *(const v8bf*)p;
  v8bf hi = *(const v8bf*)(p + 16);
  return combine16(lo, hi);
}

// B fragment (32x16 bf16) from a transposed [N][K] source:
// lane n = lane&15; lane-half selects K blocks {0..7,16..23} vs {8..15,24..31}.
static __device__ __forceinline__ v16bf load_b_frag(const ushort_t* base,
                                                    int rowStride, int col,
                                                    int k, int lane) {
  const ushort_t* p =
      base + (size_t)(col + (lane & 15)) * rowStride + k + ((lane >> 4) * 8);
  v8bf lo = *(const v8bf*)p;
  v8bf hi = *(const v8bf*)(p + 16);
  return combine16(lo, hi);
}

// ---------------------------------------------------------------------------
// Prep kernels: fp32 -> bf16 convert / transpose
// ---------------------------------------------------------------------------
__global__ __launch_bounds__(256) void cvt_bf16(const float* __restrict__ in,
                                                ushort_t* __restrict__ out,
                                                int n) {
  int i = blockIdx.x * blockDim.x + threadIdx.x;
  if (i < n) out[i] = f2bf(in[i]);
}

// w[K][N] -> wT[N][K] in bf16 (output-coalesced)
__global__ __launch_bounds__(256) void transpose_bf16(
    const float* __restrict__ w, ushort_t* __restrict__ wT, int K, int N) {
  int i = blockIdx.x * blockDim.x + threadIdx.x;
  if (i < K * N) {
    int nn = i / K;
    int kk = i - nn * K;
    wT[i] = f2bf(w[(size_t)kk * N + nn]);
  }
}

// ---------------------------------------------------------------------------
// GEMM 1: qkv = x @ w_attn + b_attn, scattered into q/k/vT head layouts.
// Block tile 128x128, 8 waves (4Mx2N), each wave 32x64 (2x4 frags).
// ---------------------------------------------------------------------------
__global__ __launch_bounds__(256) void gemm_qkv(
    const ushort_t* __restrict__ xb, const ushort_t* __restrict__ waT,
    const float* __restrict__ b_attn, ushort_t* __restrict__ qb,
    ushort_t* __restrict__ kb, ushort_t* __restrict__ vT) {
  const int lane = threadIdx.x & 31;
  const int wave = threadIdx.x >> 5;
  const int rowBase = blockIdx.y * 128 + (wave >> 1) * 32;
  const int colBase = blockIdx.x * 128 + (wave & 1) * 64;

  v8f acc[2][4];
#pragma unroll
  for (int i = 0; i < 2; ++i)
#pragma unroll
    for (int j = 0; j < 4; ++j) acc[i][j] = zero8();

  for (int kk = 0; kk < EMB; kk += 32) {
    v16bf a0 = load_a_frag(xb, EMB, rowBase, kk, lane);
    v16bf a1 = load_a_frag(xb, EMB, rowBase + 16, kk, lane);
    v16bf b0 = load_b_frag(waT, EMB, colBase, kk, lane);
    v16bf b1 = load_b_frag(waT, EMB, colBase + 16, kk, lane);
    v16bf b2 = load_b_frag(waT, EMB, colBase + 32, kk, lane);
    v16bf b3 = load_b_frag(waT, EMB, colBase + 48, kk, lane);
    acc[0][0] = wmma_bf16(a0, b0, acc[0][0]);
    acc[0][1] = wmma_bf16(a0, b1, acc[0][1]);
    acc[0][2] = wmma_bf16(a0, b2, acc[0][2]);
    acc[0][3] = wmma_bf16(a0, b3, acc[0][3]);
    acc[1][0] = wmma_bf16(a1, b0, acc[1][0]);
    acc[1][1] = wmma_bf16(a1, b1, acc[1][1]);
    acc[1][2] = wmma_bf16(a1, b2, acc[1][2]);
    acc[1][3] = wmma_bf16(a1, b3, acc[1][3]);
  }

  // Epilogue: column c -> head = c/192, within-head slice c%192 (q|k|v).
#pragma unroll
  for (int mf = 0; mf < 2; ++mf) {
#pragma unroll
    for (int nf = 0; nf < 4; ++nf) {
      const int c = colBase + nf * 16 + (lane & 15);
      const int head = c / 192;
      const int w = c - head * 192;
      const float bias = b_attn[c];
      const int rbase = rowBase + mf * 16 + (lane >> 4) * 8;
#pragma unroll
      for (int e = 0; e < 8; ++e) {
        const int row = rbase + e;           // row = b*SEQ + t
        const int b = row >> 10;
        const int t = row & (SEQ - 1);
        const int bh = b * N_HEAD + head;
        const ushort_t us = f2bf(acc[mf][nf][e] + bias);
        if (w < 64)
          qb[((size_t)bh * SEQ + t) * HEAD_DIM + w] = us;
        else if (w < 128)
          kb[((size_t)bh * SEQ + t) * HEAD_DIM + (w - 64)] = us;
        else
          vT[((size_t)bh * HEAD_DIM + (w - 128)) * SEQ + t] = us;  // transposed
      }
    }
  }
}

// ---------------------------------------------------------------------------
// Flash attention: 1 block per (b,h, 128-row q tile); each wave owns 16 rows.
// Streaming softmax over 32-wide KV tiles with causal masking.
// ---------------------------------------------------------------------------
__global__ __launch_bounds__(256) void flash_attn(
    const ushort_t* __restrict__ qb, const ushort_t* __restrict__ kb,
    const ushort_t* __restrict__ vT, ushort_t* __restrict__ ao) {
  __shared__ ushort_t Plds[8][16 * 32];  // per-wave 16x32 bf16 transpose buffer

  const int lane = threadIdx.x & 31;
  const int wave = threadIdx.x >> 5;
  const int lane15 = lane & 15;
  const int laneHalf = lane >> 4;
  const int bh = blockIdx.x >> 3;
  const int tq = blockIdx.x & 7;
  const int q0 = tq * 128 + wave * 16;
  const int b = bh >> 4, h = bh & 15;

  const ushort_t* qbase = qb + (size_t)bh * SEQ * HEAD_DIM;
  const ushort_t* kbase = kb + (size_t)bh * SEQ * HEAD_DIM;
  const ushort_t* vbase = vT + (size_t)bh * HEAD_DIM * SEQ;

  // Q as A-fragments over K = head_dim (two 32-wide chunks).
  v16bf aq0 = load_a_frag(qbase, HEAD_DIM, q0, 0, lane);
  v16bf aq1 = load_a_frag(qbase, HEAD_DIM, q0, 32, lane);

  v8f o[4];
#pragma unroll
  for (int df = 0; df < 4; ++df) o[df] = zero8();
  float rmax[8], rsum[8];
#pragma unroll
  for (int e = 0; e < 8; ++e) { rmax[e] = -3.0e38f; rsum[e] = 0.0f; }

  const float scale = 0.125f;  // 1/sqrt(64)
  const int kv_last = q0 + 15; // causal: only KV tiles intersecting the lower triangle

  for (int kv = 0; kv <= kv_last; kv += 32) {
    // S tile (16 rows x 32 kv cols) = Q · K^T, k rows act as B columns.
    v16bf bk00 = load_b_frag(kbase, HEAD_DIM, kv, 0, lane);
    v16bf bk01 = load_b_frag(kbase, HEAD_DIM, kv, 32, lane);
    v16bf bk10 = load_b_frag(kbase, HEAD_DIM, kv + 16, 0, lane);
    v16bf bk11 = load_b_frag(kbase, HEAD_DIM, kv + 16, 32, lane);
    v8f s0 = zero8(), s1 = zero8();
    s0 = wmma_bf16(aq0, bk00, s0);
    s0 = wmma_bf16(aq1, bk01, s0);
    s1 = wmma_bf16(aq0, bk10, s1);
    s1 = wmma_bf16(aq1, bk11, s1);

    const int c0 = kv + lane15;
    const int c1 = kv + 16 + lane15;
    float p0[8], p1[8], tmax[8], tsum[8], corr[8];
#pragma unroll
    for (int e = 0; e < 8; ++e) {
      const int t = q0 + laneHalf * 8 + e;
      float v0 = s0[e] * scale; if (c0 > t) v0 = -1.0e9f;  // ref: masked -> -1e9
      float v1 = s1[e] * scale; if (c1 > t) v1 = -1.0e9f;
      p0[e] = v0; p1[e] = v1;
      tmax[e] = fmaxf(v0, v1);
    }
    // Row-wise reductions: rows live across the 16 lanes of each half.
#pragma unroll
    for (int off = 1; off < 16; off <<= 1)
#pragma unroll
      for (int e = 0; e < 8; ++e)
        tmax[e] = fmaxf(tmax[e], __shfl_xor(tmax[e], off, 32));
#pragma unroll
    for (int e = 0; e < 8; ++e) {
      const float nm = fmaxf(rmax[e], tmax[e]);
      corr[e] = __expf(rmax[e] - nm);
      rmax[e] = nm;
      p0[e] = __expf(p0[e] - nm);
      p1[e] = __expf(p1[e] - nm);
      tsum[e] = p0[e] + p1[e];
    }
#pragma unroll
    for (int off = 1; off < 16; off <<= 1)
#pragma unroll
      for (int e = 0; e < 8; ++e) tsum[e] += __shfl_xor(tsum[e], off, 32);
#pragma unroll
    for (int e = 0; e < 8; ++e) rsum[e] = rsum[e] * corr[e] + tsum[e];
#pragma unroll
    for (int df = 0; df < 4; ++df)
#pragma unroll
      for (int e = 0; e < 8; ++e) o[df][e] *= corr[e];

    // D-layout (lane=col, elem=row) -> row-major LDS -> A-layout reload.
    ushort_t* pw = &Plds[wave][(laneHalf * 8) * 32 + lane15];
#pragma unroll
    for (int e = 0; e < 8; ++e) {
      pw[e * 32] = f2bf(p0[e]);
      pw[e * 32 + 16] = f2bf(p1[e]);
    }
    asm volatile("s_wait_dscnt 0" ::: "memory");  // wave-private LDS RAW fence
    const ushort_t* pr = &Plds[wave][lane15 * 32 + laneHalf * 8];
    v8bf plo = *(const v8bf*)pr;
    v8bf phi = *(const v8bf*)(pr + 16);
    v16bf pa = combine16(plo, phi);

    // O += P · V using transposed V so B-fragments are two contiguous 16B loads.
#pragma unroll
    for (int df = 0; df < 4; ++df) {
      v16bf bv = load_b_frag(vbase, SEQ, df * 16, kv, lane);
      o[df] = wmma_bf16(pa, bv, o[df]);
    }
  }

#pragma unroll
  for (int e = 0; e < 8; ++e) rsum[e] = 1.0f / rsum[e];
#pragma unroll
  for (int df = 0; df < 4; ++df) {
    const int d = df * 16 + lane15;
#pragma unroll
    for (int e = 0; e < 8; ++e) {
      const int t = q0 + laneHalf * 8 + e;
      // out[b][t][h*64+d] (bf16) == post-transpose reshape in the reference
      ao[((size_t)(b * SEQ + t)) * EMB + h * HEAD_DIM + d] =
          f2bf(o[df][e] * rsum[e]);
    }
  }
}

// ---------------------------------------------------------------------------
// GEMM 2: out = att_out @ w_proj + b_proj (fp32 output)
// ---------------------------------------------------------------------------
__global__ __launch_bounds__(256) void gemm_proj(
    const ushort_t* __restrict__ ab, const ushort_t* __restrict__ wpT,
    const float* __restrict__ b_proj, float* __restrict__ out) {
  const int lane = threadIdx.x & 31;
  const int wave = threadIdx.x >> 5;
  const int rowBase = blockIdx.y * 128 + (wave >> 1) * 32;
  const int colBase = blockIdx.x * 128 + (wave & 1) * 64;

  v8f acc[2][4];
#pragma unroll
  for (int i = 0; i < 2; ++i)
#pragma unroll
    for (int j = 0; j < 4; ++j) acc[i][j] = zero8();

  for (int kk = 0; kk < EMB; kk += 32) {
    v16bf a0 = load_a_frag(ab, EMB, rowBase, kk, lane);
    v16bf a1 = load_a_frag(ab, EMB, rowBase + 16, kk, lane);
    v16bf b0 = load_b_frag(wpT, EMB, colBase, kk, lane);
    v16bf b1 = load_b_frag(wpT, EMB, colBase + 16, kk, lane);
    v16bf b2 = load_b_frag(wpT, EMB, colBase + 32, kk, lane);
    v16bf b3 = load_b_frag(wpT, EMB, colBase + 48, kk, lane);
    acc[0][0] = wmma_bf16(a0, b0, acc[0][0]);
    acc[0][1] = wmma_bf16(a0, b1, acc[0][1]);
    acc[0][2] = wmma_bf16(a0, b2, acc[0][2]);
    acc[0][3] = wmma_bf16(a0, b3, acc[0][3]);
    acc[1][0] = wmma_bf16(a1, b0, acc[1][0]);
    acc[1][1] = wmma_bf16(a1, b1, acc[1][1]);
    acc[1][2] = wmma_bf16(a1, b2, acc[1][2]);
    acc[1][3] = wmma_bf16(a1, b3, acc[1][3]);
  }

#pragma unroll
  for (int mf = 0; mf < 2; ++mf) {
#pragma unroll
    for (int nf = 0; nf < 4; ++nf) {
      const int c = colBase + nf * 16 + (lane & 15);
      const float bias = b_proj[c];
      const int rbase = rowBase + mf * 16 + (lane >> 4) * 8;
#pragma unroll
      for (int e = 0; e < 8; ++e)
        out[(size_t)(rbase + e) * EMB + c] = acc[mf][nf][e] + bias;
    }
  }
}

// ---------------------------------------------------------------------------
extern "C" void kernel_launch(void* const* d_in, const int* in_sizes, int n_in,
                              void* d_out, int out_size, void* d_ws,
                              size_t ws_size, hipStream_t stream) {
  (void)in_sizes; (void)n_in; (void)out_size; (void)ws_size;
  const float* x      = (const float*)d_in[0];
  const float* w_attn = (const float*)d_in[1];
  const float* b_attn = (const float*)d_in[2];
  const float* w_proj = (const float*)d_in[3];
  const float* b_proj = (const float*)d_in[4];
  float* out = (float*)d_out;

  ushort_t* ws = (ushort_t*)d_ws;
  size_t off = 0;
  ushort_t* xb  = ws + off; off += (size_t)BATCH * SEQ * EMB;            // x bf16
  ushort_t* waT = ws + off; off += (size_t)QKV_N * EMB;                  // w_attn^T bf16
  ushort_t* wpT = ws + off; off += (size_t)EMB * EMB;                    // w_proj^T bf16
  ushort_t* qb  = ws + off; off += (size_t)BATCH * N_HEAD * SEQ * HEAD_DIM;
  ushort_t* kb  = ws + off; off += (size_t)BATCH * N_HEAD * SEQ * HEAD_DIM;
  ushort_t* vT  = ws + off; off += (size_t)BATCH * N_HEAD * SEQ * HEAD_DIM;
  ushort_t* ab  = ws + off; off += (size_t)BATCH * SEQ * EMB;            // attn out bf16

  cvt_bf16<<<(BATCH * SEQ * EMB) / 256, 256, 0, stream>>>(x, xb,
                                                          BATCH * SEQ * EMB);
  transpose_bf16<<<(EMB * QKV_N) / 256, 256, 0, stream>>>(w_attn, waT, EMB,
                                                          QKV_N);
  transpose_bf16<<<(EMB * EMB) / 256, 256, 0, stream>>>(w_proj, wpT, EMB, EMB);

  gemm_qkv<<<dim3(QKV_N / 128, (BATCH * SEQ) / 128), 256, 0, stream>>>(
      xb, waT, b_attn, qb, kb, vT);

  flash_attn<<<dim3(BATCH * N_HEAD * (SEQ / 128)), 256, 0, stream>>>(qb, kb,
                                                                     vT, ab);

  gemm_proj<<<dim3(EMB / 128, (BATCH * SEQ) / 128), 256, 0, stream>>>(
      ab, wpT, b_proj, out);
}
